// Model_19241453486459
// MI455X (gfx1250) — matrix operand order repaired and checked by reading, not compiled, not run on hardware
//
#include <hip/hip_runtime.h>
#include <cstdint>

// ---------------------------------------------------------------------------
// Problem dimensions
// ---------------------------------------------------------------------------
#define BATCH 512
#define LIN   260
#define H1LEN 130      // conv1 output length
#define C1CH  64
#define T2LEN 65       // conv2 output length
#define C2CH  128
#define KDIM  8320     // 128*65 flattened latent
#define NEMB  8192
#define NC    16

typedef __bf16 bf16_t;
typedef bf16_t v16bf __attribute__((ext_vector_type(16)));
typedef float  v8f   __attribute__((ext_vector_type(8)));

union Frag {
  uint4 u[2];
  v16bf v;
};

__device__ __forceinline__ unsigned short f2bf(float f) {
  unsigned u = __float_as_uint(f);
  u += 0x7fffu + ((u >> 16) & 1u);   // round-to-nearest-even
  return (unsigned short)(u >> 16);
}

#define LDS_OFF(p) ((unsigned)(unsigned long long)(__attribute__((address_space(3))) void*)(p))

__device__ __forceinline__ void async_cp32(unsigned ldsoff, const void* gaddr) {
  // one instruction moves 16B; offset: applies to BOTH lds and global address
  asm volatile("global_load_async_to_lds_b128 %0, %1, off"
               :: "v"(ldsoff), "v"(gaddr) : "memory");
  asm volatile("global_load_async_to_lds_b128 %0, %1, off offset:16"
               :: "v"(ldsoff), "v"(gaddr) : "memory");
}

// ---------------------------------------------------------------------------
// Zero scratch accumulators (counts histogram + sse scalar)
// ---------------------------------------------------------------------------
__global__ void zero_kernel(float* __restrict__ counts, float* __restrict__ sse) {
  int i = blockIdx.x * 256 + threadIdx.x;
  if (i < NEMB) counts[i] = 0.f;
  if (i == 0) sse[0] = 0.f;
}

// ---------------------------------------------------------------------------
// conv1: x (512,260) -> h (512,64,130), k=4 s=2 p=1, relu
// ---------------------------------------------------------------------------
__global__ void conv1_kernel(const float* __restrict__ x, const float* __restrict__ w,
                             const float* __restrict__ bias, float* __restrict__ h) {
  int i = blockIdx.x * 256 + threadIdx.x;
  if (i >= BATCH * C1CH * H1LEN) return;
  int t = i % H1LEN;
  int c = (i / H1LEN) & (C1CH - 1);
  int b = i / (H1LEN * C1CH);
  float acc = bias[c];
  const float* xr = x + (size_t)b * LIN;
#pragma unroll
  for (int k = 0; k < 4; ++k) {
    int p = 2 * t - 1 + k;
    if (p >= 0 && p < LIN) acc += w[c * 4 + k] * xr[p];
  }
  h[i] = fmaxf(acc, 0.f);
}

// ---------------------------------------------------------------------------
// conv2: h (512,64,130) -> z_e fp32 (512,128,65) and z bf16 (512,8320)
// ---------------------------------------------------------------------------
__global__ void conv2_kernel(const float* __restrict__ h, const float* __restrict__ w,
                             const float* __restrict__ bias, float* __restrict__ ze,
                             unsigned short* __restrict__ zb) {
  int i = blockIdx.x * 256 + threadIdx.x;
  if (i >= BATCH * C2CH * T2LEN) return;
  int t = i % T2LEN;
  int co = (i / T2LEN) & (C2CH - 1);
  int b = i / (T2LEN * C2CH);
  float acc = bias[co];
  const float* hb = h + (size_t)b * C1CH * H1LEN;
  const float* wr = w + co * 256;
  for (int ci = 0; ci < C1CH; ++ci) {
    const float* hc = hb + ci * H1LEN;
#pragma unroll
    for (int k = 0; k < 4; ++k) {
      int p = 2 * t - 1 + k;
      if (p >= 0 && p < H1LEN) acc += wr[ci * 4 + k] * hc[p];
    }
  }
  ze[i] = acc;          // i == b*8320 + co*65 + t
  zb[i] = f2bf(acc);
}

// ---------------------------------------------------------------------------
// codebook prep: fp32 (8192,8320) -> bf16 copy (L2-resident, 136MB) + row norms
// ---------------------------------------------------------------------------
__global__ void cbprep_kernel(const float* __restrict__ cb, unsigned short* __restrict__ cbb,
                              float* __restrict__ cnorm) {
  __shared__ float red[256];
  int j = blockIdx.x;
  const float* row = cb + (size_t)j * KDIM;
  unsigned short* out = cbb + (size_t)j * KDIM;
  float s = 0.f;
  for (int k = threadIdx.x; k < KDIM; k += 256) {
    float v = row[k];
    out[k] = f2bf(v);
    s += v * v;
  }
  red[threadIdx.x] = s;
  __syncthreads();
  for (int st = 128; st > 0; st >>= 1) {
    if (threadIdx.x < st) red[threadIdx.x] += red[threadIdx.x + st];
    __syncthreads();
  }
  if (threadIdx.x == 0) cnorm[j] = red[0];
}

// ---------------------------------------------------------------------------
// WMMA GEMM: dot(512,8192) = z_bf16(512x8320) @ cb_bf16^T
// Block tile 64(M) x 256(N), 8 waves, each wave 64x32 (4x2 WMMA tiles).
// A tiles (64x64 bf16 = 8KB) double-buffered in LDS via async copies; the
// async issue for tile i+1 overlaps the 16 WMMAs of tile i.  Outer loop is
// kept NOT unrolled so the 64 accumulator VGPRs stay pinned (no v_mov /
// hazard-nop churn from register-pressure spills).
// ---------------------------------------------------------------------------
__global__ __launch_bounds__(256) void gemm_dot_kernel(
    const unsigned short* __restrict__ Zb,
    const unsigned short* __restrict__ Cb,
    float* __restrict__ dotp) {
  __shared__ unsigned short As[2][64 * 64];   // 2 x 8KB A tiles, row stride 64

  const int m0 = blockIdx.x * 64;
  const int n0 = blockIdx.y * 256;
  const int tid = threadIdx.x;
  const int lane = tid & 31;
  const int wave = tid >> 5;
  const int nw = n0 + wave * 32;
  const int rl16 = lane & 15;
  const int kh = lane >> 4;

  v8f acc[4][2] = {};

  // A staging: 256 threads x 32B = 8KB per 64-K tile
  const int ar = tid >> 2;            // row 0..63
  const int ac = (tid & 3) * 16;      // bf16 col offset 0/16/32/48
  const unsigned lds0 = LDS_OFF(&As[0][ar * 64 + ac]);
  const unsigned lds1 = LDS_OFF(&As[1][ar * 64 + ac]);
  const unsigned short* grow = Zb + (size_t)(m0 + ar) * KDIM + ac;

  const int NITER = KDIM / 64;        // 130

  // prologue: stage tile 0 into buffer 0
  async_cp32(lds0, grow);

#pragma unroll 1
  for (int i = 0; i < NITER; ++i) {
    const int k0 = i * 64;
    const int cur = i & 1;

    if (i + 1 < NITER) {
      // stage next tile into the other buffer; hidden behind this tile's WMMAs
      async_cp32(cur ? lds0 : lds1, grow + (k0 + 64));
#if __has_builtin(__builtin_amdgcn_s_wait_asynccnt)
      __builtin_amdgcn_s_wait_asynccnt(2);   // wait only for CURRENT tile (in-order)
#else
      asm volatile("s_wait_asynccnt 0x2" ::: "memory");
#endif
    } else {
#if __has_builtin(__builtin_amdgcn_s_wait_asynccnt)
      __builtin_amdgcn_s_wait_asynccnt(0);
#else
      asm volatile("s_wait_asynccnt 0x0" ::: "memory");
#endif
    }
    __syncthreads();    // current tile visible to all waves

    const unsigned short* tile = &As[cur][0];
#pragma unroll
    for (int ks = 0; ks < 64; ks += 32) {
      // A fragments per ISA 16-bit A 16x32 layout:
      // lane l: row=l&15, VGPR0..3 = K kh*8+[0..7], VGPR4..7 = K 16+kh*8+[0..7]
      Frag a[4];
#pragma unroll
      for (int mt = 0; mt < 4; ++mt) {
        const unsigned short* ap = tile + (mt * 16 + rl16) * 64 + ks + kh * 8;
        a[mt].u[0] = *(const uint4*)(ap);
        a[mt].u[1] = *(const uint4*)(ap + 16);
      }
      // B fragments: lane l holds column (l&15), K = k0+ks + kh*8+[0..7], 16+kh*8+[0..7]
#pragma unroll
      for (int nt = 0; nt < 2; ++nt) {
        const unsigned short* bp =
            Cb + (size_t)(nw + nt * 16 + rl16) * KDIM + k0 + ks + kh * 8;
        Frag bf;
        bf.u[0] = *(const uint4*)(bp);
        bf.u[1] = *(const uint4*)(bp + 16);
#pragma unroll
        for (int mt = 0; mt < 4; ++mt) {
          acc[mt][nt] = __builtin_amdgcn_wmma_f32_16x16x32_bf16(
              false, a[mt].v, false, bf.v, (short)0, acc[mt][nt], false, false);
        }
      }
    }
    __syncthreads();    // all waves done reading buf[cur] before it is refilled
  }

  // C/D layout: VGPR r -> (M=r, N=lane) lanes 0-15; (M=r+8, N=lane-16) lanes 16-31
#pragma unroll
  for (int mt = 0; mt < 4; ++mt) {
#pragma unroll
    for (int nt = 0; nt < 2; ++nt) {
      int col = nw + nt * 16 + rl16;
#pragma unroll
      for (int r = 0; r < 8; ++r) {
        int row = m0 + mt * 16 + ((lane < 16) ? r : r + 8);
        dotp[(size_t)row * NEMB + col] = acc[mt][nt][r];
      }
    }
  }
}

// ---------------------------------------------------------------------------
// argmin over j of (||c_j||^2 - 2*dot[b][j])   (||z_b||^2 is row-constant)
// ---------------------------------------------------------------------------
__global__ void argmin_kernel(const float* __restrict__ dotp,
                              const float* __restrict__ cnorm,
                              int* __restrict__ idx) {
  __shared__ float sv[256];
  __shared__ int si[256];
  int b = blockIdx.x;
  float bv = 3.4e38f;
  int bi = 0x7fffffff;
  for (int j = threadIdx.x; j < NEMB; j += 256) {
    float v = cnorm[j] - 2.0f * dotp[(size_t)b * NEMB + j];
    if (v < bv || (v == bv && j < bi)) { bv = v; bi = j; }
  }
  sv[threadIdx.x] = bv;
  si[threadIdx.x] = bi;
  __syncthreads();
  for (int s = 128; s > 0; s >>= 1) {
    if (threadIdx.x < s) {
      float ov = sv[threadIdx.x + s];
      int oi = si[threadIdx.x + s];
      if (ov < sv[threadIdx.x] || (ov == sv[threadIdx.x] && oi < si[threadIdx.x])) {
        sv[threadIdx.x] = ov;
        si[threadIdx.x] = oi;
      }
    }
    __syncthreads();
  }
  if (threadIdx.x == 0) idx[b] = si[0];
}

// ---------------------------------------------------------------------------
// sum((zq - z_e)^2) + counts histogram
// ---------------------------------------------------------------------------
__global__ void vqstats_kernel(const float* __restrict__ cb, const float* __restrict__ ze,
                               const int* __restrict__ idx, float* __restrict__ sse,
                               float* __restrict__ counts) {
  __shared__ float red[256];
  int b = blockIdx.x;
  const float* c = cb + (size_t)idx[b] * KDIM;
  const float* z = ze + (size_t)b * KDIM;
  float s = 0.f;
  for (int k = threadIdx.x; k < KDIM; k += 256) {
    float d = c[k] - z[k];
    s += d * d;
  }
  red[threadIdx.x] = s;
  __syncthreads();
  for (int st = 128; st > 0; st >>= 1) {
    if (threadIdx.x < st) red[threadIdx.x] += red[threadIdx.x + st];
    __syncthreads();
  }
  if (threadIdx.x == 0) {
    atomicAdd(sse, red[0]);
    atomicAdd(&counts[idx[b]], 1.0f);
  }
}

// ---------------------------------------------------------------------------
// emb_loss + perplexity
// ---------------------------------------------------------------------------
__global__ void finalize_kernel(const float* __restrict__ counts, const float* __restrict__ sse,
                                const float* __restrict__ beta, float* __restrict__ emb,
                                float* __restrict__ perp) {
  __shared__ float red[256];
  float s = 0.f;
  for (int j = threadIdx.x; j < NEMB; j += 256) {
    float e = counts[j] * (1.f / 512.f);
    s += e * logf(e + 1e-10f);
  }
  red[threadIdx.x] = s;
  __syncthreads();
  for (int st = 128; st > 0; st >>= 1) {
    if (threadIdx.x < st) red[threadIdx.x] += red[threadIdx.x + st];
    __syncthreads();
  }
  if (threadIdx.x == 0) {
    perp[0] = expf(-red[0]);
    emb[0] = (1.f + beta[0]) * sse[0] / (float)(BATCH * KDIM);
  }
}

// ---------------------------------------------------------------------------
// small GEMM heads: one wave per output dot product
// ---------------------------------------------------------------------------
__global__ void mt_kernel(const float* __restrict__ ze, const float* __restrict__ w,
                          const float* __restrict__ bias, float* __restrict__ out) {
  int lane = threadIdx.x & 31;
  int oidx = blockIdx.x * 8 + (threadIdx.x >> 5);
  int b = oidx >> 4, o = oidx & 15;
  const float* zr = ze + (size_t)b * KDIM;            // first 64 ch = first 4160
  const float* wr = w + (size_t)o * 4160;
  float acc = 0.f;
  for (int k = lane; k < 4160; k += 32) acc += zr[k] * wr[k];
  for (int s = 16; s > 0; s >>= 1) acc += __shfl_xor(acc, s, 32);
  if (lane == 0) out[b * NC + o] = acc + bias[o];
}

__global__ void adv1_kernel(const float* __restrict__ ze, const float* __restrict__ w,
                            const float* __restrict__ bias, float* __restrict__ out) {
  int lane = threadIdx.x & 31;
  int oidx = blockIdx.x * 8 + (threadIdx.x >> 5);
  int b = oidx >> 8, o = oidx & 255;
  const float* zr = ze + (size_t)b * KDIM + 4160;     // channels 64..127
  const float* wr = w + (size_t)o * 4160;
  float acc = 0.f;
  for (int k = lane; k < 4160; k += 32) acc += zr[k] * wr[k];
  for (int s = 16; s > 0; s >>= 1) acc += __shfl_xor(acc, s, 32);
  if (lane == 0) out[b * 256 + o] = fmaxf(acc + bias[o], 0.f);
}

__global__ void adv2_kernel(const float* __restrict__ a1, const float* __restrict__ w,
                            const float* __restrict__ bias, float* __restrict__ out) {
  int lane = threadIdx.x & 31;
  int oidx = blockIdx.x * 8 + (threadIdx.x >> 5);
  int b = oidx >> 7, o = oidx & 127;
  const float* ar = a1 + (size_t)b * 256;
  const float* wr = w + (size_t)o * 256;
  float acc = 0.f;
  for (int k = lane; k < 256; k += 32) acc += ar[k] * wr[k];
  for (int s = 16; s > 0; s >>= 1) acc += __shfl_xor(acc, s, 32);
  if (lane == 0) out[b * 128 + o] = fmaxf(acc + bias[o], 0.f);
}

__global__ void adv3_kernel(const float* __restrict__ a2, const float* __restrict__ w,
                            const float* __restrict__ bias, float* __restrict__ out) {
  int lane = threadIdx.x & 31;
  int oidx = blockIdx.x * 8 + (threadIdx.x >> 5);
  int b = oidx >> 4, o = oidx & 15;
  const float* ar = a2 + (size_t)b * 128;
  const float* wr = w + (size_t)o * 128;
  float acc = 0.f;
  for (int k = lane; k < 128; k += 32) acc += ar[k] * wr[k];
  for (int s = 16; s > 0; s >>= 1) acc += __shfl_xor(acc, s, 32);
  if (lane == 0) out[b * NC + o] = acc + bias[o];
}

// ---------------------------------------------------------------------------
// decoder: transposed convs as gather.  z_q value == codebook[idx[b]]
// ---------------------------------------------------------------------------
__global__ void dec1_kernel(const float* __restrict__ cb, const int* __restrict__ idx,
                            const float* __restrict__ w, const float* __restrict__ bias,
                            float* __restrict__ hd) {
  int i = blockIdx.x * 256 + threadIdx.x;
  if (i >= BATCH * C1CH * H1LEN) return;
  int t = i % H1LEN;
  int co = (i / H1LEN) & (C1CH - 1);
  int b = i / (H1LEN * C1CH);
  const float* zq = cb + (size_t)idx[b] * KDIM;
  int p0 = (t + 1) & 1;              // valid k parities: p0 and p0+2
  int s0 = (t + 1 - p0) >> 1;
  int s1 = (t - 1 - p0) >> 1;
  bool v0 = (s0 >= 0 && s0 < T2LEN);
  bool v1 = (s1 >= 0 && s1 < T2LEN);
  float acc = bias[co];
  for (int ci = 0; ci < C2CH; ++ci) {
    const float* zc = zq + ci * T2LEN;
    const float* wc = w + (ci * C1CH + co) * 4;
    if (v0) acc += wc[p0] * zc[s0];
    if (v1) acc += wc[p0 + 2] * zc[s1];
  }
  hd[i] = fmaxf(acc, 0.f);
}

__global__ void dec2_kernel(const float* __restrict__ hd, const float* __restrict__ w,
                            const float* __restrict__ bias, float* __restrict__ xhat) {
  int i = blockIdx.x * 256 + threadIdx.x;
  if (i >= BATCH * LIN) return;
  int t = i % LIN;
  int b = i / LIN;
  int p0 = (t + 1) & 1;
  int s0 = (t + 1 - p0) >> 1;
  int s1 = (t - 1 - p0) >> 1;
  bool v0 = (s0 >= 0 && s0 < H1LEN);
  bool v1 = (s1 >= 0 && s1 < H1LEN);
  float acc = bias[0];
  const float* hb = hd + (size_t)b * C1CH * H1LEN;
  for (int ci = 0; ci < C1CH; ++ci) {
    const float* hc = hb + ci * H1LEN;
    if (v0) acc += w[ci * 4 + p0] * hc[s0];
    if (v1) acc += w[ci * 4 + p0 + 2] * hc[s1];
  }
  xhat[i] = acc;
}

// ---------------------------------------------------------------------------
// launch
// ---------------------------------------------------------------------------
extern "C" void kernel_launch(void* const* d_in, const int* in_sizes, int n_in,
                              void* d_out, int out_size, void* d_ws, size_t ws_size,
                              hipStream_t stream) {
  (void)in_sizes; (void)n_in; (void)out_size; (void)ws_size;
  const float* x    = (const float*)d_in[0];
  const float* cbf  = (const float*)d_in[1];
  const float* c1w  = (const float*)d_in[2];
  const float* c1b  = (const float*)d_in[3];
  const float* c2w  = (const float*)d_in[4];
  const float* c2b  = (const float*)d_in[5];
  const float* d1w  = (const float*)d_in[6];
  const float* d1b  = (const float*)d_in[7];
  const float* d2w  = (const float*)d_in[8];
  const float* d2b  = (const float*)d_in[9];
  const float* mtw  = (const float*)d_in[10];
  const float* mtb  = (const float*)d_in[11];
  const float* aw1  = (const float*)d_in[12];
  const float* ab1  = (const float*)d_in[13];
  const float* aw2  = (const float*)d_in[14];
  const float* ab2  = (const float*)d_in[15];
  const float* aw3  = (const float*)d_in[16];
  const float* ab3  = (const float*)d_in[17];
  const float* beta = (const float*)d_in[18];

  char* ws = (char*)d_ws;
  size_t off = 0;
  auto alloc = [&](size_t bytes) -> char* {
    char* p = ws + off;
    off += (bytes + 255) & ~(size_t)255;
    return p;
  };
  float*          h      = (float*)alloc((size_t)BATCH * C1CH * H1LEN * 4);
  float*          ze     = (float*)alloc((size_t)BATCH * KDIM * 4);
  unsigned short* zb     = (unsigned short*)alloc((size_t)BATCH * KDIM * 2);
  unsigned short* cbb    = (unsigned short*)alloc((size_t)NEMB * KDIM * 2);
  float*          cnorm  = (float*)alloc((size_t)NEMB * 4);
  float*          dotp   = (float*)alloc((size_t)BATCH * NEMB * 4);
  int*            idx    = (int*)alloc((size_t)BATCH * 4);
  float*          counts = (float*)alloc((size_t)NEMB * 4);
  float*          sse    = (float*)alloc(256);
  float*          a1     = (float*)alloc((size_t)BATCH * 256 * 4);
  float*          a2     = (float*)alloc((size_t)BATCH * 128 * 4);
  float*          hd     = (float*)alloc((size_t)BATCH * C1CH * H1LEN * 4);

  float* out    = (float*)d_out;
  float* embout = out;                       // [0]
  float* xhat   = out + 1;                   // 512*260
  float* mtout  = out + 1 + BATCH * LIN;     // 512*16
  float* advout = mtout + BATCH * NC;        // 512*16
  float* perp   = advout + BATCH * NC;       // [last]

  zero_kernel<<<32, 256, 0, stream>>>(counts, sse);
  conv1_kernel<<<16640, 256, 0, stream>>>(x, c1w, c1b, h);
  conv2_kernel<<<16640, 256, 0, stream>>>(h, c2w, c2b, ze, zb);
  cbprep_kernel<<<NEMB, 256, 0, stream>>>(cbf, cbb, cnorm);

  dim3 ggrid(BATCH / 64, NEMB / 256);
  gemm_dot_kernel<<<ggrid, 256, 0, stream>>>(zb, cbb, dotp);

  argmin_kernel<<<BATCH, 256, 0, stream>>>(dotp, cnorm, idx);
  vqstats_kernel<<<BATCH, 256, 0, stream>>>(cbf, ze, idx, sse, counts);
  finalize_kernel<<<1, 256, 0, stream>>>(counts, sse, beta, embout, perp);

  mt_kernel<<<1024, 256, 0, stream>>>(ze, mtw, mtb, mtout);
  adv1_kernel<<<16384, 256, 0, stream>>>(ze, aw1, ab1, a1);
  adv2_kernel<<<8192, 256, 0, stream>>>(a1, aw2, ab2, a2);
  adv3_kernel<<<1024, 256, 0, stream>>>(a2, aw3, ab3, advout);

  dec1_kernel<<<16640, 256, 0, stream>>>(cbf, idx, d1w, d1b, hd);
  dec2_kernel<<<520, 256, 0, stream>>>(hd, d2w, d2b, xhat);
}
